// DGLInteractionNetwork_7653631722048
// MI455X (gfx1250) — matrix operand order repaired and checked by reading, not compiled, not run on hardware
//
#include <hip/hip_runtime.h>

typedef __attribute__((ext_vector_type(16))) __bf16 v16bf;
typedef __attribute__((ext_vector_type(8)))  float  v8f;

#define DD   64      // node/edge feature dim
#define HIDN 128     // hidden dim
#define K3D  192     // edge MLP input dim
#define TM   128     // rows (edges/nodes) per workgroup tile

// ---------- helpers ----------
__device__ __forceinline__ unsigned pk_bf16(float a, float b) {
  unsigned ua = __float_as_uint(a), ub = __float_as_uint(b);
  ua = (ua + 0x7FFFu + ((ua >> 16) & 1u)) >> 16;   // RNE
  ub = (ub + 0x7FFFu + ((ub >> 16) & 1u)) >> 16;
  return (ua & 0xFFFFu) | (ub << 16);
}
__device__ __forceinline__ unsigned short bf16_1(float a) {
  unsigned ua = __float_as_uint(a);
  return (unsigned short)((ua + 0x7FFFu + ((ua >> 16) & 1u)) >> 16);
}
__device__ __forceinline__ v16bf as_v16bf(uint4 x, uint4 y) {
  union { uint4 q[2]; v16bf v; } u;
  u.q[0] = x; u.q[1] = y;
  return u.v;
}
__device__ __forceinline__ v8f v8f_zero() {
  v8f z = {0.f,0.f,0.f,0.f,0.f,0.f,0.f,0.f};
  return z;
}
// A fragment (16x32 bf16). Rows across lanes 0-15; dword d holds K pair
// kk = laneHi*4 + (d&3) + (d>=4)*8  -> two 16B loads at +0 and +8 dwords.
__device__ __forceinline__ v16bf load_a_frag(const unsigned* base, int rowDw, int ksDw, int laneHi) {
  const unsigned* p = base + rowDw + ksDw + (laneHi << 2);
  return as_v16bf(*(const uint4*)p, *(const uint4*)(p + 8));
}
// B fragment (32x16 bf16). Cols across lanes 0-15; dword d holds K pair
// kk = laneHi*4 + (d&3) + (d>=4)*8  -> two 16B loads at +0 and +8 dwords.
__device__ __forceinline__ v16bf load_b_frag(const unsigned* base, int rowDw, int ksDw, int laneHi) {
  const unsigned* p = base + rowDw + ksDw + (laneHi << 2);
  return as_v16bf(*(const uint4*)p, *(const uint4*)(p + 8));
}
#define WMMA_BF16(a,b,c) \
  __builtin_amdgcn_wmma_f32_16x16x32_bf16(false,(a),false,(b),(short)0,(c),false,false)

// Native non-returning f32 atomic add at DEVICE scope (RMW resolves in L2;
// the 25.6MB agg buffer is L2-resident). Guarantees global_atomic_add_f32
// instead of a possible CAS-loop expansion of the C++ atomic builtin.
__device__ __forceinline__ void atomic_add_f32(float* p, float v) {
  asm volatile("global_atomic_add_f32 %0, %1, off scope:SCOPE_DEV"
               :
               : "v"(p), "v"(v)
               : "memory");
}

// ---------- kernel 0: zero agg/output ----------
__global__ __launch_bounds__(256) void gn_zero_f4(float4* __restrict__ p, int n4) {
  int i = blockIdx.x * 256 + threadIdx.x;
  if (i < n4) p[i] = make_float4(0.f, 0.f, 0.f, 0.f);
}

// ---------- kernel 1: pack all weights into transposed bf16-pair layout ----------
// ws dword layout: [0,12288) W1t(K=192,N=128) | [12288,16384) W2t(K=128,N=64)
//                  [16384,24576) Wn1t(K=128,N=128) | [24576,28672) Wn2t(K=128,N=64)
__global__ __launch_bounds__(256) void gn_pack_weights(
    const float* __restrict__ We1, const float* __restrict__ We2,
    const float* __restrict__ Wn1, const float* __restrict__ Wn2,
    unsigned* __restrict__ ws) {
  int t = blockIdx.x * 256 + threadIdx.x;
  const float* W; int Kd, N, off;
  if      (t < 12288) { W = We1; Kd = 96; N = 128; off = 0;     }
  else if (t < 16384) { W = We2; Kd = 64; N = 64;  off = 12288; }
  else if (t < 24576) { W = Wn1; Kd = 64; N = 128; off = 16384; }
  else if (t < 28672) { W = Wn2; Kd = 64; N = 64;  off = 24576; }
  else return;
  int d  = t - off;
  int n  = d / Kd;
  int kk = d - n * Kd;
  ws[t] = pk_bf16(W[(size_t)(2*kk) * N + n], W[(size_t)(2*kk+1) * N + n]);
}

// ---------- kernel 2: edge block (gather + MLP + scatter-add) ----------
__global__ __launch_bounds__(256) void gn_edge_kernel(
    const float* __restrict__ node_feat, const float* __restrict__ edge_feat,
    const int* __restrict__ senders, const int* __restrict__ receivers,
    const float* __restrict__ be1, const float* __restrict__ be2,
    const unsigned* __restrict__ W1t, const unsigned* __restrict__ W2t,
    float* __restrict__ agg, int nEdges) {
  // sU: A tile bf16-packed [128][96] (48KB); reused as H bf16 [128][128] (32KB)
  __shared__ unsigned sU[TM * (K3D / 2)];
  __shared__ int sRecv[TM];
  __shared__ int sSend[TM];

  const int tid   = threadIdx.x;
  const int eBase = blockIdx.x * TM;

  if (tid < TM) {
    int e = eBase + tid;
    sRecv[tid] = (e < nEdges) ? receivers[e] : 0;
  } else {
    int e = eBase + (tid - TM);
    sSend[tid - TM] = (e < nEdges) ? senders[e] : 0;
  }
  __syncthreads();

  // Stage A: concat([edge, node[recv], node[send]]) -> packed bf16
  for (int idx = tid; idx < TM * (K3D / 2); idx += 256) {
    int row = idx / (K3D / 2);
    int c   = idx - row * (K3D / 2);
    int e   = eBase + row;
    float2 v = make_float2(0.f, 0.f);
    if (e < nEdges) {
      if      (c < 32) v = *(const float2*)(edge_feat + (size_t)e * DD + 2 * c);
      else if (c < 64) v = *(const float2*)(node_feat + (size_t)sRecv[row] * DD + 2 * (c - 32));
      else             v = *(const float2*)(node_feat + (size_t)sSend[row] * DD + 2 * (c - 64));
    }
    sU[idx] = pk_bf16(v.x, v.y);
  }
  __syncthreads();

  const int lane   = tid & 31;
  const int wv     = tid >> 5;        // 8 waves: wave w owns M-strip [16w,16w+16)
  const int laneHi = lane >> 4;
  const int laneLo = lane & 15;
  const int mBase  = wv * 16;

  // GEMM1: [128x192] x [192x128]
  v8f acc[8];
#pragma unroll
  for (int i = 0; i < 8; ++i) acc[i] = v8f_zero();
#pragma unroll
  for (int ks = 0; ks < K3D / 32; ++ks) {
    v16bf a = load_a_frag(sU, (mBase + laneLo) * (K3D / 2), ks * 16, laneHi);
#pragma unroll
    for (int nt = 0; nt < 8; ++nt) {
      v16bf b = load_b_frag(W1t, (nt * 16 + laneLo) * (K3D / 2), ks * 16, laneHi);
      acc[nt] = WMMA_BF16(a, b, acc[nt]);
    }
  }
  __syncthreads();                    // A tile dead -> reuse sU for H

  // bias + ReLU -> H bf16 [128][128]
  unsigned short* sH = (unsigned short*)sU;
#pragma unroll
  for (int nt = 0; nt < 8; ++nt) {
    int   n  = nt * 16 + laneLo;
    float b1 = be1[n];
#pragma unroll
    for (int r = 0; r < 8; ++r) {
      int   m = mBase + r + 8 * laneHi;
      float h = fmaxf(acc[nt][r] + b1, 0.f);
      sH[m * HIDN + n] = bf16_1(h);
    }
  }
  __syncthreads();

  // GEMM2: [128x128] x [128x64]
  v8f acc2[4];
#pragma unroll
  for (int i = 0; i < 4; ++i) acc2[i] = v8f_zero();
#pragma unroll
  for (int ks = 0; ks < HIDN / 32; ++ks) {
    v16bf a = load_a_frag(sU, (mBase + laneLo) * (HIDN / 2), ks * 16, laneHi);
#pragma unroll
    for (int nt = 0; nt < 4; ++nt) {
      v16bf b = load_b_frag(W2t, (nt * 16 + laneLo) * (HIDN / 2), ks * 16, laneHi);
      acc2[nt] = WMMA_BF16(a, b, acc2[nt]);
    }
  }

  // bias + scatter-add into agg (native f32 global atomics, L2-resident)
#pragma unroll
  for (int nt = 0; nt < 4; ++nt) {
    int   col = nt * 16 + laneLo;
    float b2  = be2[col];
#pragma unroll
    for (int r = 0; r < 8; ++r) {
      int m = mBase + r + 8 * laneHi;
      int e = eBase + m;
      if (e < nEdges)
        atomic_add_f32(agg + (size_t)sRecv[m] * DD + col, acc2[nt][r] + b2);
    }
  }
}

// ---------- kernel 3: node block (out holds agg on entry, n_out on exit) ----------
__global__ __launch_bounds__(256) void gn_node_kernel(
    const float* __restrict__ node_feat,
    const float* __restrict__ bn1, const float* __restrict__ bn2,
    const unsigned* __restrict__ Wn1t, const unsigned* __restrict__ Wn2t,
    float* __restrict__ out, int nNodes) {
  // sU: A tile bf16 [128][64] (32KB); reused as H bf16 [128][128] (32KB)
  __shared__ unsigned sU[TM * (HIDN / 2)];

  const int tid  = threadIdx.x;
  const int base = blockIdx.x * TM;

  // Stage A: concat([agg(=out), node_feat]) -> packed bf16
  for (int idx = tid; idx < TM * (HIDN / 2); idx += 256) {
    int row = idx >> 6;
    int c   = idx & 63;
    int nd  = base + row;
    float2 v = make_float2(0.f, 0.f);
    if (nd < nNodes) {
      if (c < 32) v = *(const float2*)(out       + (size_t)nd * DD + 2 * c);
      else        v = *(const float2*)(node_feat + (size_t)nd * DD + 2 * (c - 32));
    }
    sU[idx] = pk_bf16(v.x, v.y);
  }
  __syncthreads();

  const int lane   = tid & 31;
  const int wv     = tid >> 5;
  const int laneHi = lane >> 4;
  const int laneLo = lane & 15;
  const int mBase  = wv * 16;

  // GEMM1: [128x128] x [128x128]
  v8f acc[8];
#pragma unroll
  for (int i = 0; i < 8; ++i) acc[i] = v8f_zero();
#pragma unroll
  for (int ks = 0; ks < HIDN / 32; ++ks) {
    v16bf a = load_a_frag(sU, (mBase + laneLo) * (HIDN / 2), ks * 16, laneHi);
#pragma unroll
    for (int nt = 0; nt < 8; ++nt) {
      v16bf b = load_b_frag(Wn1t, (nt * 16 + laneLo) * (HIDN / 2), ks * 16, laneHi);
      acc[nt] = WMMA_BF16(a, b, acc[nt]);
    }
  }
  __syncthreads();

  unsigned short* sH = (unsigned short*)sU;
#pragma unroll
  for (int nt = 0; nt < 8; ++nt) {
    int   n  = nt * 16 + laneLo;
    float b1 = bn1[n];
#pragma unroll
    for (int r = 0; r < 8; ++r) {
      int   m = mBase + r + 8 * laneHi;
      float h = fmaxf(acc[nt][r] + b1, 0.f);
      sH[m * HIDN + n] = bf16_1(h);
    }
  }
  __syncthreads();

  // GEMM2: [128x128] x [128x64]
  v8f acc2[4];
#pragma unroll
  for (int i = 0; i < 4; ++i) acc2[i] = v8f_zero();
#pragma unroll
  for (int ks = 0; ks < HIDN / 32; ++ks) {
    v16bf a = load_a_frag(sU, (mBase + laneLo) * (HIDN / 2), ks * 16, laneHi);
#pragma unroll
    for (int nt = 0; nt < 4; ++nt) {
      v16bf b = load_b_frag(Wn2t, (nt * 16 + laneLo) * (HIDN / 2), ks * 16, laneHi);
      acc2[nt] = WMMA_BF16(a, b, acc2[nt]);
    }
  }

#pragma unroll
  for (int nt = 0; nt < 4; ++nt) {
    int   col = nt * 16 + laneLo;
    float b2  = bn2[col];
#pragma unroll
    for (int r = 0; r < 8; ++r) {
      int m  = mBase + r + 8 * laneHi;
      int nd = base + m;
      if (nd < nNodes)
        out[(size_t)nd * DD + col] = acc2[nt][r] + b2;
    }
  }
}

// ---------- launcher ----------
extern "C" void kernel_launch(void* const* d_in, const int* in_sizes, int n_in,
                              void* d_out, int out_size, void* d_ws, size_t ws_size,
                              hipStream_t stream) {
  const float* node_feat = (const float*)d_in[0];
  const float* edge_feat = (const float*)d_in[1];
  const int*   senders   = (const int*)  d_in[2];
  const int*   receivers = (const int*)  d_in[3];
  const float* We1 = (const float*)d_in[4];
  const float* be1 = (const float*)d_in[5];
  const float* We2 = (const float*)d_in[6];
  const float* be2 = (const float*)d_in[7];
  const float* Wn1 = (const float*)d_in[8];
  const float* bn1 = (const float*)d_in[9];
  const float* Wn2 = (const float*)d_in[10];
  const float* bn2 = (const float*)d_in[11];

  const int nNodes = in_sizes[0] / DD;
  const int nEdges = in_sizes[2];

  float*    outp = (float*)d_out;       // doubles as agg buffer
  unsigned* ws   = (unsigned*)d_ws;     // packed bf16 weights (~112 KB)
  const unsigned* W1t  = ws;
  const unsigned* W2t  = ws + 12288;
  const unsigned* Wn1t = ws + 16384;
  const unsigned* Wn2t = ws + 24576;

  int n4 = nNodes * DD / 4;
  gn_zero_f4<<<(n4 + 255) / 256, 256, 0, stream>>>((float4*)outp, n4);
  gn_pack_weights<<<112, 256, 0, stream>>>(We1, We2, Wn1, Wn2, ws);
  gn_edge_kernel<<<(nEdges + TM - 1) / TM, 256, 0, stream>>>(
      node_feat, edge_feat, senders, receivers, be1, be2, W1t, W2t, outp, nEdges);
  gn_node_kernel<<<(nNodes + TM - 1) / TM, 256, 0, stream>>>(
      node_feat, bn1, bn2, Wn1t, Wn2t, outp, nNodes);
}
